// FourierBlock_70411693850728
// MI455X (gfx1250) — matrix operand order repaired and checked by reading, not compiled, not run on hardware
//
#include <hip/hip_runtime.h>
#include <hip/hip_bf16.h>

// ---------------- problem constants ----------------
#define B_ 4
#define S_ 4096
#define D_ 1024
#define H_ 16
#define HD_ 64
#define NF_ 2049        // rfft bins for S=4096
#define NFC_ 256
#define M_TOT (B_ * S_) // 16384 rows for every GEMM

#define FFT_N 4096
#define FFT_LOG2 12

typedef __attribute__((ext_vector_type(16))) _Float16 v16h;
typedef __attribute__((ext_vector_type(8)))  _Float16 v8h;
typedef __attribute__((ext_vector_type(8)))  float    v8f;
typedef __attribute__((ext_vector_type(4)))  int      v4i;

// =====================================================================
// CDNA5 async copy (GLOBAL_LOAD_ASYNC_TO_LDS_B128, tracked by ASYNCcnt)
// with a safe fallback when the builtin is absent.
// Probe-learned signature: (int4 AS1* src, int4 AS3* dst, imm off, imm cpol)
// =====================================================================
#if __has_builtin(__builtin_amdgcn_global_load_async_to_lds_b128) && \
    __has_builtin(__builtin_amdgcn_s_wait_asynccnt)
#define USE_ASYNC_LDS 1
typedef __attribute__((address_space(1))) v4i as1_v4i;
typedef __attribute__((address_space(3))) v4i as3_v4i;
#endif

__device__ __forceinline__ void cp16_to_lds(const _Float16* g, _Float16* l) {
#ifdef USE_ASYNC_LDS
  __builtin_amdgcn_global_load_async_to_lds_b128((as1_v4i*)g, (as3_v4i*)l, 0, 0);
#else
  *(float4*)l = *(const float4*)g;
#endif
}

__device__ __forceinline__ void async_wait_all() {
#ifdef USE_ASYNC_LDS
  __builtin_amdgcn_s_wait_asynccnt(0);
#endif
}

// =====================================================================
// fp32 -> fp16 pre-conversion (amortized across all GEMM re-reads)
// =====================================================================
__global__ void __launch_bounds__(256)
f32_to_f16_kernel(const float* __restrict__ src, _Float16* __restrict__ dst, int n4) {
  int i = blockIdx.x * 256 + threadIdx.x;
  if (i >= n4) return;
  float4 x = *(const float4*)(src + (size_t)i * 4);
  _Float16* d = dst + (size_t)i * 4;
  d[0] = (_Float16)x.x; d[1] = (_Float16)x.y;
  d[2] = (_Float16)x.z; d[3] = (_Float16)x.w;
}

// =====================================================================
// WMMA GEMM: C[M,N] = A[M,K](f16) * W[N,K](f16)^T + bias[N]
// Block = 8 waves = 128x64 tile, double-buffered LDS staging (async),
// wave = 32x32, K-step 64 -> 8 x v_wmma_f32_16x16x32_f16 per step.
// =====================================================================
#define BM 128
#define BN 64
#define BK 64
#define LDSTRIDE 72   // padded row stride in halves (144B, 16B aligned)

// A fragment (16x32 f16): lane group g: halves 0..7 = K[kk+8g..], 8..15 = K[kk+16+8g..]
__device__ __forceinline__ v16h ld_a_frag(const _Float16* p, int g) {
  v8h lo = *(const v8h*)(p + (g << 3));
  v8h hi = *(const v8h*)(p + 16 + (g << 3));
  v16h f;
#pragma unroll
  for (int i = 0; i < 8; ++i) { f[i] = lo[i]; f[8 + i] = hi[i]; }
  return f;
}

// B fragment (32x16 f16): lane group g: halves 0..7 = K[kk+8g..], 8..15 = K[kk+16+8g..]
__device__ __forceinline__ v16h ld_b_frag(const _Float16* p, int g) {
  v8h lo = *(const v8h*)(p + (g << 3));
  v8h hi = *(const v8h*)(p + 16 + (g << 3));
  v16h f;
#pragma unroll
  for (int i = 0; i < 8; ++i) { f[i] = lo[i]; f[8 + i] = hi[i]; }
  return f;
}

__global__ void __launch_bounds__(256)
gemm_bias_wmma(const _Float16* __restrict__ A, const _Float16* __restrict__ W,
               const float* __restrict__ bias, float* __restrict__ C,
               int M, int N, int K) {
  __shared__ _Float16 Ab[2][BM * LDSTRIDE];
  __shared__ _Float16 Bb[2][BN * LDSTRIDE];

  const int tid  = threadIdx.x;
  const int lane = tid & 31;
  const int wave = tid >> 5;
  const int wm = wave & 3;             // 4 waves along M
  const int wn = wave >> 2;            // 2 waves along N
  const int mblk = blockIdx.x * BM;
  const int nblk = blockIdx.y * BN;
  const int lr = lane & 15;
  const int g  = lane >> 4;

  v8f acc00 = {}; v8f acc01 = {}; v8f acc10 = {}; v8f acc11 = {};

  auto stageA = [&](int buf, int k0) {
#pragma unroll
    for (int u = tid; u < (BM * BK) / 8; u += 256) {       // 1024 16B units
      int row = u >> 3, c = (u & 7) << 3;
      cp16_to_lds(A + (size_t)(mblk + row) * K + k0 + c,
                  &Ab[buf][row * LDSTRIDE + c]);
    }
  };
  auto stageB = [&](int buf, int k0) {
#pragma unroll
    for (int u = tid; u < (BN * BK) / 8; u += 256) {       // 512 16B units
      int row = u >> 3, c = (u & 7) << 3;
      cp16_to_lds(W + (size_t)(nblk + row) * K + k0 + c,
                  &Bb[buf][row * LDSTRIDE + c]);
    }
  };

  stageA(0, 0);
  stageB(0, 0);
  async_wait_all();
  __syncthreads();

  int buf = 0;
  for (int k0 = 0; k0 < K; k0 += BK) {
    if (k0 + BK < K) {                 // overlap next tile copy with compute
      stageA(buf ^ 1, k0 + BK);
      stageB(buf ^ 1, k0 + BK);
    }
#pragma unroll
    for (int kk = 0; kk < BK; kk += 32) {
      const _Float16* arow0 = &Ab[buf][(wm * 32 + lr)      * LDSTRIDE + kk];
      const _Float16* arow1 = &Ab[buf][(wm * 32 + 16 + lr) * LDSTRIDE + kk];
      const _Float16* brow0 = &Bb[buf][(wn * 32 + lr)      * LDSTRIDE + kk];
      const _Float16* brow1 = &Bb[buf][(wn * 32 + 16 + lr) * LDSTRIDE + kk];
      v16h a0 = ld_a_frag(arow0, g);
      v16h a1 = ld_a_frag(arow1, g);
      v16h b0 = ld_b_frag(brow0, g);
      v16h b1 = ld_b_frag(brow1, g);
      acc00 = __builtin_amdgcn_wmma_f32_16x16x32_f16(false, a0, false, b0, (short)0, acc00, false, false);
      acc01 = __builtin_amdgcn_wmma_f32_16x16x32_f16(false, a0, false, b1, (short)0, acc01, false, false);
      acc10 = __builtin_amdgcn_wmma_f32_16x16x32_f16(false, a1, false, b0, (short)0, acc10, false, false);
      acc11 = __builtin_amdgcn_wmma_f32_16x16x32_f16(false, a1, false, b1, (short)0, acc11, false, false);
    }
    async_wait_all();
    __syncthreads();
    buf ^= 1;
  }

  // C/D layout: lane (col n = lr, group g), VGPR r holds row M = 8g + r
  const int m0 = mblk + wm * 32;
  const int n0 = nblk + wn * 32;
  const int n_a = n0 + lr;
  const int n_b = n0 + 16 + lr;
  const float bv_a = bias[n_a];
  const float bv_b = bias[n_b];
#pragma unroll
  for (int r = 0; r < 8; ++r) {
    int mA = m0 + g * 8 + r;
    int mB = m0 + 16 + g * 8 + r;
    C[(size_t)mA * N + n_a] = acc00[r] + bv_a;
    C[(size_t)mA * N + n_b] = acc01[r] + bv_b;
    C[(size_t)mB * N + n_a] = acc10[r] + bv_a;
    C[(size_t)mB * N + n_b] = acc11[r] + bv_b;
  }
}

// =====================================================================
// Forward FFT: one block per channel (b,h,hd); grid.y selects q/k/v.
// Radix-2 DIT in LDS, real input, writes bins 0..2048 to (B,H,F,HD).
// =====================================================================
__global__ void __launch_bounds__(256)
fft_fwd_kernel(const float* __restrict__ Qp, const float* __restrict__ Kp,
               const float* __restrict__ Vp,
               float2* __restrict__ qf, float2* __restrict__ kf,
               float2* __restrict__ vf) {
  __shared__ float2 buf[FFT_N];
  const int c    = blockIdx.x;
  const int tsel = blockIdx.y;
  const float* src = (tsel == 0) ? Qp : (tsel == 1) ? Kp : Vp;
  float2* dst      = (tsel == 0) ? qf : (tsel == 1) ? kf : vf;

  const int b  = c >> 10;
  const int h  = (c >> 6) & 15;
  const int hd = c & 63;
  const float* x = src + ((size_t)b * S_) * D_ + h * HD_ + hd;  // stride D_

  for (int i = threadIdx.x; i < FFT_N; i += 256) {
    int r = __brev((unsigned)i) >> (32 - FFT_LOG2);
    buf[r] = make_float2(x[(size_t)i * D_], 0.0f);
  }
  __syncthreads();

  for (int half = 1; half < FFT_N; half <<= 1) {
    const float step = -3.14159265358979323846f / (float)half;
    for (int t = threadIdx.x; t < FFT_N / 2; t += 256) {
      int pos = t & (half - 1);
      int i0  = ((t & ~(half - 1)) << 1) | pos;
      int i1  = i0 + half;
      float sn, cs;
      __sincosf(step * (float)pos, &sn, &cs);
      float2 u = buf[i0], w = buf[i1];
      float2 tw = make_float2(cs * w.x - sn * w.y, cs * w.y + sn * w.x);
      buf[i0] = make_float2(u.x + tw.x, u.y + tw.y);
      buf[i1] = make_float2(u.x - tw.x, u.y - tw.y);
    }
    __syncthreads();
  }

  const size_t base = ((size_t)(b * H_ + h)) * NF_;
  for (int f = threadIdx.x; f < NF_; f += 256)
    dst[(base + f) * HD_ + hd] = buf[f];
}

// =====================================================================
// mag[b,h,f] = mean_hd |xf|   (grid.y selects tensor)
// =====================================================================
__global__ void mag_kernel(const float2* __restrict__ qf,
                           const float2* __restrict__ kf,
                           const float2* __restrict__ vf,
                           float* __restrict__ mag) {
  const int total = B_ * H_ * NF_;
  int t = blockIdx.x * blockDim.x + threadIdx.x;
  if (t >= total) return;
  const int tsel = blockIdx.y;
  const float2* src = (tsel == 0) ? qf : (tsel == 1) ? kf : vf;
  const float2* p = src + (size_t)t * HD_;
  float s = 0.0f;
#pragma unroll 4
  for (int hd = 0; hd < HD_; ++hd) {
    float2 v = p[hd];
    s += sqrtf(v.x * v.x + v.y * v.y);
  }
  mag[(size_t)tsel * total + t] = s * (1.0f / HD_);
}

// =====================================================================
// Exact top-256 (rank-ordered, lowest-index tie break) per (b,h) row.
// =====================================================================
__global__ void __launch_bounds__(256)
topk_kernel(const float* __restrict__ mag, int* __restrict__ idx_out) {
  __shared__ float m[NF_];
  __shared__ float rv[256];
  __shared__ int   ri[256];
  const int bh = blockIdx.x;
  const int tsel = blockIdx.y;
  const float* row = mag + ((size_t)tsel * (B_ * H_) + bh) * NF_;
  int* out = idx_out + ((size_t)tsel * (B_ * H_) + bh) * NFC_;

  for (int f = threadIdx.x; f < NF_; f += 256) m[f] = row[f];
  __syncthreads();

  for (int r = 0; r < NFC_; ++r) {
    float best = -3.0e38f;
    int   bi   = NF_;
    for (int f = threadIdx.x; f < NF_; f += 256) {
      float v = m[f];
      if (v > best || (v == best && f < bi)) { best = v; bi = f; }
    }
    rv[threadIdx.x] = best;
    ri[threadIdx.x] = bi;
    __syncthreads();
    for (int s = 128; s > 0; s >>= 1) {
      if (threadIdx.x < s) {
        float ov = rv[threadIdx.x + s];
        int   oi = ri[threadIdx.x + s];
        if (ov > rv[threadIdx.x] ||
            (ov == rv[threadIdx.x] && oi < ri[threadIdx.x])) {
          rv[threadIdx.x] = ov; ri[threadIdx.x] = oi;
        }
      }
      __syncthreads();
    }
    if (threadIdx.x == 0) {
      out[r] = ri[0];
      m[ri[0]] = -3.0e38f;
    }
    __syncthreads();
  }
}

__global__ void zero_spec_kernel(float2* __restrict__ p, int n) {
  int t = blockIdx.x * blockDim.x + threadIdx.x;
  if (t < n) p[t] = make_float2(0.0f, 0.0f);
}

// =====================================================================
// out_full[b,h,q_idx[r],hd] = qf[q_idx]*conj(kf[k_idx])*vf[v_idx] * w[h,r,hd]
// =====================================================================
__global__ void combine_kernel(const float2* __restrict__ qf,
                               const float2* __restrict__ kf,
                               const float2* __restrict__ vf,
                               const int* __restrict__ idx,
                               const float* __restrict__ fw,
                               float2* __restrict__ outfull) {
  const int tid = blockIdx.x * blockDim.x + threadIdx.x;  // B*H*NFC*HD = 1M
  const int hd = tid & 63;
  const int r  = (tid >> 6) & 255;
  const int bh = tid >> 14;
  const int h  = bh & 15;

  const int qi = idx[((size_t)0 * (B_ * H_) + bh) * NFC_ + r];
  const int ki = idx[((size_t)1 * (B_ * H_) + bh) * NFC_ + r];
  const int vi = idx[((size_t)2 * (B_ * H_) + bh) * NFC_ + r];

  const size_t rowbase = (size_t)bh * NF_;
  float2 a = qf[(rowbase + qi) * HD_ + hd];
  float2 b = kf[(rowbase + ki) * HD_ + hd];
  float2 c = vf[(rowbase + vi) * HD_ + hd];

  float2 t1 = make_float2(a.x * b.x + a.y * b.y, a.y * b.x - a.x * b.y);   // a*conj(b)
  float2 t2 = make_float2(t1.x * c.x - t1.y * c.y, t1.x * c.y + t1.y * c.x);
  float wr = fw[(((size_t)0 * H_ + h) * NFC_ + r) * HD_ + hd];
  float wi = fw[(((size_t)1 * H_ + h) * NFC_ + r) * HD_ + hd];
  float2 o = make_float2(t2.x * wr - t2.y * wi, t2.x * wi + t2.y * wr);

  outfull[(rowbase + qi) * HD_ + hd] = o;
}

// =====================================================================
// Inverse FFT: Hermitian-extend, complex iFFT, write f16 directly into
// the (B,S,D) merged-head layout (feeds the final WMMA GEMM).
// =====================================================================
__global__ void __launch_bounds__(256)
fft_inv_kernel(const float2* __restrict__ spec, _Float16* __restrict__ out) {
  __shared__ float2 buf[FFT_N];
  const int c  = blockIdx.x;
  const int b  = c >> 10;
  const int h  = (c >> 6) & 15;
  const int hd = c & 63;
  const float2* X = spec + ((size_t)(b * H_ + h)) * NF_ * HD_ + hd;  // f stride HD_

  for (int i = threadIdx.x; i < FFT_N; i += 256) {
    float2 v;
    if (i <= FFT_N / 2) {
      v = X[(size_t)i * HD_];
    } else {
      float2 t = X[(size_t)(FFT_N - i) * HD_];
      v = make_float2(t.x, -t.y);
    }
    int r = __brev((unsigned)i) >> (32 - FFT_LOG2);
    buf[r] = v;
  }
  __syncthreads();

  for (int half = 1; half < FFT_N; half <<= 1) {
    const float step = 3.14159265358979323846f / (float)half;  // +i for inverse
    for (int t = threadIdx.x; t < FFT_N / 2; t += 256) {
      int pos = t & (half - 1);
      int i0  = ((t & ~(half - 1)) << 1) | pos;
      int i1  = i0 + half;
      float sn, cs;
      __sincosf(step * (float)pos, &sn, &cs);
      float2 u = buf[i0], w = buf[i1];
      float2 tw = make_float2(cs * w.x - sn * w.y, cs * w.y + sn * w.x);
      buf[i0] = make_float2(u.x + tw.x, u.y + tw.y);
      buf[i1] = make_float2(u.x - tw.x, u.y - tw.y);
    }
    __syncthreads();
  }

  const float scale = 1.0f / (float)FFT_N;
  for (int s = threadIdx.x; s < FFT_N; s += 256)
    out[((size_t)b * S_ + s) * D_ + h * HD_ + hd] = (_Float16)(buf[s].x * scale);
}

// =====================================================================
// launcher
// =====================================================================
extern "C" void kernel_launch(void* const* d_in, const int* in_sizes, int n_in,
                              void* d_out, int out_size, void* d_ws, size_t ws_size,
                              hipStream_t stream) {
  const float* q  = (const float*)d_in[0];
  const float* k  = (const float*)d_in[1];
  const float* v  = (const float*)d_in[2];
  const float* Wq = (const float*)d_in[3];
  const float* bq = (const float*)d_in[4];
  const float* Wk = (const float*)d_in[5];
  const float* bk = (const float*)d_in[6];
  const float* Wv = (const float*)d_in[7];
  const float* bv = (const float*)d_in[8];
  const float* Wo = (const float*)d_in[9];
  const float* bo = (const float*)d_in[10];
  const float* fw = (const float*)d_in[11];

  const size_t SZ_PROJ   = (size_t)M_TOT * D_ * sizeof(float);            // 64 MiB
  const size_t SZ_PROJ_H = (size_t)M_TOT * D_ * sizeof(_Float16);         // 32 MiB
  const size_t SZ_SPEC   = (size_t)B_ * H_ * NF_ * HD_ * sizeof(float2);  // ~64 MiB
  const size_t SZ_MAG    = (size_t)3 * B_ * H_ * NF_ * sizeof(float);
  const size_t SZ_IDX    = (size_t)3 * B_ * H_ * NFC_ * sizeof(int);

  char* w = (char*)d_ws;
  // region A: Qp/Kp/Vp (f32) during proj+FFT; later reused for the scattered
  // spectrum (outfull @0) and the f16 time-domain attn (@ SZ_SPEC).
  float*     Qp      = (float*)(w);
  float*     Kp      = (float*)(w + SZ_PROJ);
  float*     Vp      = (float*)(w + 2 * SZ_PROJ);
  float2*    outfull = (float2*)(w);
  _Float16*  attn_h  = (_Float16*)(w + SZ_SPEC);
  // region B: spectra; first reused for the f16 copies of q/k/v.
  char*      base2 = w + 3 * SZ_PROJ;
  _Float16*  qh = (_Float16*)(base2);
  _Float16*  kh = (_Float16*)(base2 + SZ_PROJ_H);
  _Float16*  vh = (_Float16*)(base2 + 2 * SZ_PROJ_H);
  float2*    qf = (float2*)(base2);
  float2*    kf = (float2*)(base2 + SZ_SPEC);
  float2*    vf = (float2*)(base2 + 2 * SZ_SPEC);
  // region C: mag, idx, f16 weights
  char*      base3 = base2 + 3 * SZ_SPEC;
  float*     mag = (float*)(base3);
  int*       idx = (int*)(base3 + SZ_MAG);
  _Float16*  Wqh = (_Float16*)(base3 + SZ_MAG + SZ_IDX);
  _Float16*  Wkh = Wqh + (size_t)D_ * D_;
  _Float16*  Wvh = Wkh + (size_t)D_ * D_;
  _Float16*  Woh = Wvh + (size_t)D_ * D_;

  // 0) one-time fp32 -> fp16 conversions
  {
    const int n4_in = (M_TOT * D_) / 4;   // 4,194,304
    const int n4_w  = (D_ * D_) / 4;      // 262,144
    f32_to_f16_kernel<<<(n4_in + 255) / 256, 256, 0, stream>>>(q,  qh,  n4_in);
    f32_to_f16_kernel<<<(n4_in + 255) / 256, 256, 0, stream>>>(k,  kh,  n4_in);
    f32_to_f16_kernel<<<(n4_in + 255) / 256, 256, 0, stream>>>(v,  vh,  n4_in);
    f32_to_f16_kernel<<<(n4_w  + 255) / 256, 256, 0, stream>>>(Wq, Wqh, n4_w);
    f32_to_f16_kernel<<<(n4_w  + 255) / 256, 256, 0, stream>>>(Wk, Wkh, n4_w);
    f32_to_f16_kernel<<<(n4_w  + 255) / 256, 256, 0, stream>>>(Wv, Wvh, n4_w);
    f32_to_f16_kernel<<<(n4_w  + 255) / 256, 256, 0, stream>>>(Wo, Woh, n4_w);
  }

  dim3 gemm_grid(M_TOT / BM, D_ / BN);

  // 1) projections (WMMA, async-LDS double buffered)
  gemm_bias_wmma<<<gemm_grid, 256, 0, stream>>>(qh, Wqh, bq, Qp, M_TOT, D_, D_);
  gemm_bias_wmma<<<gemm_grid, 256, 0, stream>>>(kh, Wkh, bk, Kp, M_TOT, D_, D_);
  gemm_bias_wmma<<<gemm_grid, 256, 0, stream>>>(vh, Wvh, bv, Vp, M_TOT, D_, D_);

  // 2) forward FFTs (q,k,v)
  fft_fwd_kernel<<<dim3(B_ * H_ * HD_, 3), 256, 0, stream>>>(Qp, Kp, Vp, qf, kf, vf);

  // 3) magnitudes + exact rank-ordered top-256
  {
    const int total = B_ * H_ * NF_;
    mag_kernel<<<dim3((total + 255) / 256, 3), 256, 0, stream>>>(qf, kf, vf, mag);
  }
  topk_kernel<<<dim3(B_ * H_, 3), 256, 0, stream>>>(mag, idx);

  // 4) zero spectrum then gather/multiply/scatter
  {
    const int nspec = B_ * H_ * NF_ * HD_;
    zero_spec_kernel<<<(nspec + 255) / 256, 256, 0, stream>>>(outfull, nspec);
  }
  combine_kernel<<<(B_ * H_ * NFC_ * HD_) / 256, 256, 0, stream>>>(qf, kf, vf, idx, fw, outfull);

  // 5) inverse FFT -> (B,S,D) merged heads, f16
  fft_inv_kernel<<<B_ * H_ * HD_, 256, 0, stream>>>(outfull, attn_h);

  // 6) output projection -> d_out (f32)
  gemm_bias_wmma<<<gemm_grid, 256, 0, stream>>>(attn_h, Woh, bo, (float*)d_out, M_TOT, D_, D_);
}
